// WindowAttentionV2_76493367542589
// MI455X (gfx1250) — matrix operand order repaired and checked by reading, not compiled, not run on hardware
//
#include <hip/hip_runtime.h>
#include <hip/hip_bf16.h>
#include <math.h>

// ---------- types ----------
typedef __bf16 bf16_t;
typedef bf16_t v16bf __attribute__((ext_vector_type(16)));
typedef float  v8f   __attribute__((ext_vector_type(8)));
typedef unsigned short u16;

union V16 { v16bf bf; u16 u[16]; unsigned u32v[8]; };

// native converts (lower to v_cvt_pk_bf16_f32)
__device__ __forceinline__ u16 f2bf(float f) {
  return __builtin_bit_cast(u16, (bf16_t)f);
}
__device__ __forceinline__ unsigned pkbf(float a, float b) {
  union { bf16_t h[2]; unsigned u; } p;
  p.h[0] = (bf16_t)a; p.h[1] = (bf16_t)b;
  return p.u;
}

#if __has_builtin(__builtin_amdgcn_exp2f)
#define EXP2(x) __builtin_amdgcn_exp2f(x)
#else
#define EXP2(x) exp2f(x)
#endif

#define WMMA_BF16(A, B, Cacc) \
  __builtin_amdgcn_wmma_f32_16x16x32_bf16(false, (A), false, (B), (short)0, (Cacc), false, false)

// Problem constants: B_=64, N=512, C=192, H=6, hd=32, nW=16, table rows = 15^3 = 3375

// ---------- kernel 1: CPB MLP table  tbl[3375][6] ----------
__global__ void k_cpb_table(const float* __restrict__ rel_table,
                            const float* __restrict__ w1, const float* __restrict__ b1,
                            const float* __restrict__ w2, float* __restrict__ tbl) {
  int row = blockIdx.x * blockDim.x + threadIdx.x;
  if (row >= 3375) return;
  float c0 = rel_table[row*3+0], c1 = rel_table[row*3+1], c2 = rel_table[row*3+2];
  float acc[6] = {0.f,0.f,0.f,0.f,0.f,0.f};
  for (int j = 0; j < 512; ++j) {
    float h = fmaf(w1[j*3+0], c0, fmaf(w1[j*3+1], c1, fmaf(w1[j*3+2], c2, b1[j])));
    h = fmaxf(h, 0.f);
    #pragma unroll
    for (int hh = 0; hh < 6; ++hh) acc[hh] = fmaf(w2[hh*512+j], h, acc[hh]);
  }
  #pragma unroll
  for (int hh = 0; hh < 6; ++hh) tbl[row*6+hh] = acc[hh];
}

// ---------- kernel 2: bias[h][n][m] = 16*sigmoid(tbl[idx[n,m]][h]), 4-wide ----------
__global__ void k_bias(const int* __restrict__ rel_index, const float* __restrict__ tbl,
                       float* __restrict__ bias) {
  int t4 = (blockIdx.x * blockDim.x + threadIdx.x) * 4;
  if (t4 >= 512*512) return;
  int4 idx = *(const int4*)(rel_index + t4);
  #pragma unroll
  for (int h = 0; h < 6; ++h) {
    float4 o;
    o.x = 16.f / (1.f + __expf(-tbl[idx.x*6 + h]));
    o.y = 16.f / (1.f + __expf(-tbl[idx.y*6 + h]));
    o.z = 16.f / (1.f + __expf(-tbl[idx.z*6 + h]));
    o.w = 16.f / (1.f + __expf(-tbl[idx.w*6 + h]));
    *(float4*)(bias + (size_t)h*512*512 + t4) = o;
  }
}

// ---------- kernel 3: QKV GEMM (bf16 WMMA) + fused cosine-norm / logit scale ----------
// Each block's 32-col slab = one head's full hd for one of q/k/v (seg = blockIdx.y/6,
// head = blockIdx.y%6 are scalar). q,k are L2-normalized in f32 in-register (cross-lane
// shfl reduction within 16-lane halves), q also scaled by clamped exp(logit_scale).
// q,k -> [B,H,N,hd] bf16 ; v -> transposed [B,H,hd,N] bf16 (direct WMMA A operand later)
__global__ void __launch_bounds__(128) k_qkv(const float* __restrict__ x,
                                             const float* __restrict__ w,
                                             const float* __restrict__ qb,
                                             const float* __restrict__ vb,
                                             const float* __restrict__ logit_scale,
                                             u16* __restrict__ qo, u16* __restrict__ ko,
                                             u16* __restrict__ vt) {
  const int wave = threadIdx.x >> 5, lane = threadIdx.x & 31;
  const int half = lane >> 4, lm = lane & 15;
  const int Mbase = blockIdx.x * 64 + wave * 16;   // rows of x (32768 total)
  const int Nbase = blockIdx.y * 32;               // cols of qkv output (576 total)
  const int seg   = blockIdx.y / 6;                // 0=q, 1=k, 2=v   (scalar)
  const int head  = blockIdx.y % 6;                // head index      (scalar)
  v8f acc0 = {}; v8f acc1 = {};
  for (int k0 = 0; k0 < 192; k0 += 32) {
    // A tile: 16x32 of x; A layout: row M = lm, K = half*16 + 0..15 (contiguous)
    V16 A;
    {
      const float* xr = x + (size_t)(Mbase + lm) * 192 + k0 + half * 16;
      float fa[16];
      ((float4*)fa)[0] = ((const float4*)xr)[0];
      ((float4*)fa)[1] = ((const float4*)xr)[1];
      ((float4*)fa)[2] = ((const float4*)(xr + 8))[0];
      ((float4*)fa)[3] = ((const float4*)(xr + 8))[1];
      #pragma unroll
      for (int e = 0; e < 16; ++e) A.bf[e] = (bf16_t)fa[e];
    }
    // B tiles: B[k][n] = w[n][k]; lane: n = lm, K = k0 + half*16 + 0..15 (contiguous)
    V16 B0, B1;
    #pragma unroll
    for (int j = 0; j < 2; ++j) {
      const float* wp = w + (size_t)(Nbase + j*16 + lm) * 192 + k0 + half * 16;
      float fb[16];
      ((float4*)fb)[0] = ((const float4*)wp)[0];
      ((float4*)fb)[1] = ((const float4*)wp)[1];
      ((float4*)fb)[2] = ((const float4*)wp)[2];
      ((float4*)fb)[3] = ((const float4*)wp)[3];
      V16& B = j ? B1 : B0;
      #pragma unroll
      for (int e = 0; e < 16; ++e) B.bf[e] = (bf16_t)fb[e];
    }
    acc0 = WMMA_BF16(A.bf, B0.bf, acc0);
    acc1 = WMMA_BF16(A.bf, B1.bf, acc1);
  }

  // qkv bias (q_bias, 0, v_bias) -- per-lane d = j*16+lm within this head
  float bias0 = 0.f, bias1 = 0.f;
  if (seg == 0) { bias0 = qb[head*32 + lm]; bias1 = qb[head*32 + 16 + lm]; }
  if (seg == 2) { bias0 = vb[head*32 + lm]; bias1 = vb[head*32 + 16 + lm]; }
  #pragma unroll
  for (int r = 0; r < 8; ++r) { acc0[r] += bias0; acc1[r] += bias1; }

  // fused cosine normalization for q/k (full hd=32 lives in this block's slab).
  // D rows: halves hold distinct rows -> reduce within 16-lane halves.
  if (seg < 2) {
    float sc = 1.f;
    if (seg == 0) sc = __expf(fminf(logit_scale[head], 4.60517018598809f)); // log(100)
    #pragma unroll
    for (int r = 0; r < 8; ++r) {
      float sq = fmaf(acc0[r], acc0[r], acc1[r] * acc1[r]);
      #pragma unroll
      for (int off = 8; off >= 1; off >>= 1) sq += __shfl_xor(sq, off, 16);
      float inv = sc / fmaxf(sqrtf(sq), 1e-12f);
      acc0[r] *= inv; acc1[r] *= inv;
    }
  }

  u16* dst = (seg == 0) ? qo : ((seg == 1) ? ko : vt);
  #pragma unroll
  for (int j = 0; j < 2; ++j) {
    v8f a = j ? acc1 : acc0;
    int d = j*16 + lm;
    #pragma unroll
    for (int r = 0; r < 8; ++r) {
      int rowg = Mbase + r + half * 8;             // D layout: M = r + 8*half
      int b = rowg >> 9, n = rowg & 511;
      size_t idx = (seg == 2)
        ? (((size_t)(b*6 + head)) * 32 + d) * 512 + n      // v: [B,H,hd,N]
        : (((size_t)(b*6 + head)) * 512 + n) * 32 + d;     // q,k: [B,H,N,hd]
      dst[idx] = f2bf(a[r]);
    }
  }
}

// ---------- kernel 4: fused flash attention (transposed form, LDS-free) ----------
// S^T = K*Q^T and O^T = V^T*P^T: the query index is the WMMA N coord (= lane%16)
// everywhere -> per-lane scalar softmax state (log2 domain), one cross-half shfl per
// reduction, P^T built by packed half-swap shuffles, contiguous vector loads/stores.
__global__ void __launch_bounds__(128) k_attn(const u16* __restrict__ qo,
                                              const u16* __restrict__ ko,
                                              const u16* __restrict__ vt,
                                              const float* __restrict__ bias,
                                              const float* __restrict__ mask,
                                              u16* __restrict__ aout) {
  const int wave = threadIdx.x >> 5, lane = threadIdx.x & 31;
  const int half = lane >> 4, lm = lane & 15;
  const int qt = blockIdx.x, h = blockIdx.y, b = blockIdx.z;
  const int bh = b * 6 + h;
  const int q0 = qt * 64 + wave * 16;
  const int qg = q0 + lm;                 // this lane's query
  const float LOG2E = 1.44269504088896f;

  // B operand: Q^T (K = d = e + 16*half, N = q = lm); loaded once
  V16 Bq;
  {
    const u16* qp = qo + ((size_t)bh * 512 + qg) * 32 + half * 16;
    ((float4*)Bq.u)[0] = ((const float4*)qp)[0];
    ((float4*)Bq.u)[1] = ((const float4*)qp)[1];
  }

  v8f o0 = {}; v8f o1 = {};               // O^T tiles: d = r+8*half (+16), col = q
  float m_q = -1e30f, l_q = 0.f;          // m_q in log2 domain

  const float* bp = bias + (size_t)h * 512 * 512 + (size_t)qg * 512;
  const float* mp = mask + (size_t)(b & 15) * 512 * 512 + (size_t)qg * 512;

  #pragma unroll 2
  for (int m0 = 0; m0 < 512; m0 += 32) {
    // A operands: K tiles (A row M = lm -> key m0+lm / m0+16+lm; K = d chunks)
    V16 Ak0, Ak1;
    {
      const u16* kp = ko + ((size_t)bh * 512 + m0 + lm) * 32 + half * 16;
      ((float4*)Ak0.u)[0] = *(const float4*)kp;
      ((float4*)Ak0.u)[1] = *(const float4*)(kp + 8);
      const u16* kp1 = kp + 16 * 32;
      ((float4*)Ak1.u)[0] = *(const float4*)kp1;
      ((float4*)Ak1.u)[1] = *(const float4*)(kp1 + 8);
    }
    v8f st0 = {}; v8f st1 = {};
    st0 = WMMA_BF16(Ak0.bf, Bq.bf, st0);  // st0[r]: key = m0 + r + 8*half
    st1 = WMMA_BF16(Ak1.bf, Bq.bf, st1);  // st1[r]: key = m0 + 16 + r + 8*half

    // bias + window mask (contiguous 8-float runs), fold LOG2E scaling in
    {
      float bb[16], mm[16];
      const float* bpp = bp + m0 + half * 8;
      const float* mpp = mp + m0 + half * 8;
      ((float4*)bb)[0] = ((const float4*)bpp)[0];
      ((float4*)bb)[1] = ((const float4*)bpp)[1];
      ((float4*)bb)[2] = ((const float4*)(bpp + 16))[0];
      ((float4*)bb)[3] = ((const float4*)(bpp + 16))[1];
      ((float4*)mm)[0] = ((const float4*)mpp)[0];
      ((float4*)mm)[1] = ((const float4*)mpp)[1];
      ((float4*)mm)[2] = ((const float4*)(mpp + 16))[0];
      ((float4*)mm)[3] = ((const float4*)(mpp + 16))[1];
      #pragma unroll
      for (int r = 0; r < 8; ++r) {
        st0[r] = (st0[r] + bb[r]     + mm[r])     * LOG2E;
        st1[r] = (st1[r] + bb[r + 8] + mm[r + 8]) * LOG2E;
      }
    }

    // online softmax in log2 domain: per-lane scalar state, one cross-half shfl
    float mx = fmaxf(st0[0], st1[0]);
    #pragma unroll
    for (int r = 1; r < 8; ++r) mx = fmaxf(mx, fmaxf(st0[r], st1[r]));
    mx = fmaxf(mx, __shfl_xor(mx, 16));
    float mnew  = fmaxf(m_q, mx);
    float alpha = EXP2(m_q - mnew);
    m_q = mnew;
    float rs = 0.f;
    #pragma unroll
    for (int r = 0; r < 8; ++r) {
      st0[r] = EXP2(st0[r] - mnew);
      st1[r] = EXP2(st1[r] - mnew);
      rs += st0[r] + st1[r];
    }
    rs += __shfl_xor(rs, 16);
    l_q = l_q * alpha + rs;
    #pragma unroll
    for (int r = 0; r < 8; ++r) { o0[r] *= alpha; o1[r] *= alpha; }

    // P^T as B operand: need (K = key = e + 16*half, N = q = lm).
    // D-layout holds keys {8*srcHalf + r} (st0) / {16 + 8*srcHalf + r} (st1):
    // a half-wave swap away. Pack adjacent pairs to bf16x2 first, then shuffle
    // packed u32s (shuffle-of-pack == pack-of-shuffles): 8 shfl instead of 16.
    V16 Bp;
    {
      unsigned pk0[4], pk1[4];
      #pragma unroll
      for (int i = 0; i < 4; ++i) {
        pk0[i] = pkbf(st0[2*i], st0[2*i+1]);
        pk1[i] = pkbf(st1[2*i], st1[2*i+1]);
      }
      #pragma unroll
      for (int i = 0; i < 4; ++i) {
        unsigned s0x = (unsigned)__shfl_xor((int)pk0[i], 16);
        unsigned s1x = (unsigned)__shfl_xor((int)pk1[i], 16);
        Bp.u32v[i]     = half ? s1x : pk0[i];   // elements e = 2i, 2i+1
        Bp.u32v[i + 4] = half ? pk1[i] : s0x;   // elements e = 8+2i, 8+2i+1
      }
    }

    // A operands: V^T tiles (row M = lm -> d = lm / 16+lm; K = key chunks)
    V16 Av0, Av1;
    {
      const u16* vp0 = vt + ((size_t)bh * 32 + lm) * 512 + m0 + half * 16;
      ((float4*)Av0.u)[0] = *(const float4*)vp0;
      ((float4*)Av0.u)[1] = *(const float4*)(vp0 + 8);
      const u16* vp1 = vt + ((size_t)bh * 32 + 16 + lm) * 512 + m0 + half * 16;
      ((float4*)Av1.u)[0] = *(const float4*)vp1;
      ((float4*)Av1.u)[1] = *(const float4*)(vp1 + 8);
    }
    o0 = WMMA_BF16(Av0.bf, Bp.bf, o0);
    o1 = WMMA_BF16(Av1.bf, Bp.bf, o1);
  }

  // finalize: two contiguous 8x bf16 vector stores per lane
  float inv = 1.f / l_q;
  union { float4 v; u16 s[8]; } t0, t1;
  #pragma unroll
  for (int r = 0; r < 8; ++r) { t0.s[r] = f2bf(o0[r] * inv); t1.s[r] = f2bf(o1[r] * inv); }
  u16* op = aout + ((size_t)b * 512 + qg) * 192 + h * 32 + half * 8;
  *(float4*)op        = t0.v;   // d = 8*half + 0..7
  *(float4*)(op + 16) = t1.v;   // d = 16 + 8*half + 0..7
}

// ---------- kernel 5: output projection GEMM (bf16 WMMA) -> f32 d_out ----------
__global__ void __launch_bounds__(128) k_proj(const u16* __restrict__ a,
                                              const float* __restrict__ w,
                                              const float* __restrict__ pb,
                                              float* __restrict__ out) {
  const int wave = threadIdx.x >> 5, lane = threadIdx.x & 31;
  const int half = lane >> 4, lm = lane & 15;
  const int Mbase = blockIdx.x * 64 + wave * 16;
  const int Nbase = blockIdx.y * 32;
  v8f acc0 = {}; v8f acc1 = {};
  for (int k0 = 0; k0 < 192; k0 += 32) {
    V16 A;
    {
      const u16* ap = a + (size_t)(Mbase + lm) * 192 + k0 + half * 16;
      ((float4*)A.u)[0] = *(const float4*)ap;
      ((float4*)A.u)[1] = *(const float4*)(ap + 8);
    }
    V16 B0, B1;
    #pragma unroll
    for (int j = 0; j < 2; ++j) {
      const float* wp = w + (size_t)(Nbase + j*16 + lm) * 192 + k0 + half * 16;
      float fb[16];
      ((float4*)fb)[0] = ((const float4*)wp)[0];
      ((float4*)fb)[1] = ((const float4*)wp)[1];
      ((float4*)fb)[2] = ((const float4*)wp)[2];
      ((float4*)fb)[3] = ((const float4*)wp)[3];
      V16& B = j ? B1 : B0;
      #pragma unroll
      for (int e = 0; e < 16; ++e) B.bf[e] = (bf16_t)fb[e];
    }
    acc0 = WMMA_BF16(A.bf, B0.bf, acc0);
    acc1 = WMMA_BF16(A.bf, B1.bf, acc1);
  }
  #pragma unroll
  for (int j = 0; j < 2; ++j) {
    v8f acc = j ? acc1 : acc0;
    int col = Nbase + j*16 + lm;
    float bb = pb[col];
    #pragma unroll
    for (int r = 0; r < 8; ++r) {
      int rowg = Mbase + r + half * 8;
      out[(size_t)rowg * 192 + col] = acc[r] + bb;
    }
  }
}

// ---------- host ----------
extern "C" void kernel_launch(void* const* d_in, const int* in_sizes, int n_in,
                              void* d_out, int out_size, void* d_ws, size_t ws_size,
                              hipStream_t stream) {
  (void)in_sizes; (void)n_in; (void)out_size; (void)ws_size;
  const float* x           = (const float*)d_in[0];
  const float* mask        = (const float*)d_in[1];
  const float* qkv_w       = (const float*)d_in[2];
  const float* q_bias      = (const float*)d_in[3];
  const float* v_bias      = (const float*)d_in[4];
  const float* logit_scale = (const float*)d_in[5];
  const float* cpb_w1      = (const float*)d_in[6];
  const float* cpb_b1      = (const float*)d_in[7];
  const float* cpb_w2      = (const float*)d_in[8];
  const float* proj_w      = (const float*)d_in[9];
  const float* proj_b      = (const float*)d_in[10];
  const float* rel_table   = (const float*)d_in[11];
  const int*   rel_index   = (const int*)d_in[12];

  const size_t QKV_ELEMS = (size_t)64 * 6 * 512 * 32;   // per q/k/v tensor (bf16)
  u16* qo = (u16*)d_ws;
  u16* ko = qo + QKV_ELEMS;
  u16* vt = ko + QKV_ELEMS;                              // v stored transposed [B,H,hd,N]
  u16* ao = vt + QKV_ELEMS;                              // [B,N,C] bf16
  float* biasbuf = (float*)(ao + (size_t)64 * 512 * 192);
  float* tbl     = biasbuf + (size_t)6 * 512 * 512;

  k_cpb_table<<<(3375 + 127) / 128, 128, 0, stream>>>(rel_table, cpb_w1, cpb_b1, cpb_w2, tbl);
  k_bias<<<(512 * 512 / 4) / 128, 128, 0, stream>>>(rel_index, tbl, biasbuf);
  k_qkv<<<dim3(512, 18), 128, 0, stream>>>(x, qkv_w, q_bias, v_bias, logit_scale, qo, ko, vt);
  k_attn<<<dim3(8, 6, 64), 128, 0, stream>>>(qo, ko, vt, biasbuf, mask, ao);
  k_proj<<<dim3(512, 6), 128, 0, stream>>>(ao, proj_w, proj_b, (float*)d_out);
}
